// SimilarityFusionEncoder_28991029248258
// MI455X (gfx1250) — matrix operand order repaired and checked by reading, not compiled, not run on hardware
//
#include <hip/hip_runtime.h>
#include <hip/hip_bf16.h>
#include <math.h>
#include <stdint.h>

// ---------------------------------------------------------------------------
// Problem constants (from reference): B=4, N=1024, C0=512, H=8, K=16
// ---------------------------------------------------------------------------
static constexpr int BB = 4;
static constexpr int NN = 1024;
static constexpr int TT = BB * NN;   // 4096 tokens, t = b*N + n
static constexpr int HH = 8;
static constexpr int KK = 16;

typedef __bf16 bf16;
typedef __attribute__((ext_vector_type(16))) __bf16 v16bf;
typedef __attribute__((ext_vector_type(8)))  float  v8f;

union FragU { uint4 q[2]; v16bf v; };

// ---------------------------------------------------------------------------
// WMMA fragment loaders, matching CDNA5 ISA VGPR layouts (CDNA5 ISA)
// A: 16x32 bf16 (MxK).  lanes 0-15: v0-3 -> K0..7, v4-7 -> K16..23 (row M=lane)
//                        lanes16-31: v0-3 -> K8..15, v4-7 -> K24..31
// Per lane: two contiguous 16B runs -> two global_load_b128.
// ---------------------------------------------------------------------------
__device__ inline v16bf load_a_frag(const bf16* __restrict__ A, int lda, int lane) {
  const int r  = lane & 15;
  const int kh = (lane >> 4) << 3;  // 0 or 8 (elements)
  const bf16* p = A + (size_t)r * lda + kh;
  FragU f;
  f.q[0] = *(const uint4*)p;          // elements kh .. kh+7
  f.q[1] = *(const uint4*)(p + 16);   // elements 16+kh .. 16+kh+7
  return f.v;
}

// B: 32x16 bf16 (KxN), B[k][c] = W[o0+c][k0+k] with W row-major [O,Kdim].
// lanes 0-15: cols, K=0..7,16..23; lanes 16-31: same cols, K=8..15,24..31.
// Per lane: two 8-bf16 runs (kh.., 16+kh..) of one W row -> two global_load_b128.
__device__ inline v16bf load_b_frag(const bf16* __restrict__ W, int ldw, int lane) {
  const int c  = lane & 15;
  const int kh = (lane >> 4) << 3;  // 0 or 8 (elements)
  const bf16* p = W + (size_t)c * ldw + kh;
  FragU f;
  f.q[0] = *(const uint4*)p;
  f.q[1] = *(const uint4*)(p + 16);
  return f.v;
}

// ---------------------------------------------------------------------------
// Generic WMMA GEMM:  Y[t,o] = act( sum_k A[t,k] * W[o,k] + bias[o] )
// Block = 256 threads = 8 waves (4 M x 2 N), block tile 128(M) x 128(O),
// wave tile 32x64 = 2x4 v_wmma_f32_16x16x32_bf16 accumulators
// (8 WMMA per 6 fragment fetches per k-step).
// Epilogue fan-out: up to two bf16 destinations (column-concat buffers),
// one f32 destination, or the [B,O,N] final-output layout.
// ---------------------------------------------------------------------------
template <int ACT, int OUT_BCN>
__global__ __launch_bounds__(256) void gemm_wmma(
    const bf16* __restrict__ A, int lda,
    const bf16* __restrict__ W, int ldw,
    const float* __restrict__ bias, int Kdim,
    bf16* __restrict__ o_b1, int ldo1,
    bf16* __restrict__ o_b2, int ldo2,
    float* __restrict__ o_f, int ldof,
    float* __restrict__ o_bcn, int Ochan, int Nn) {
  const int lane = threadIdx.x & 31;
  const int wave = threadIdx.x >> 5;
  const int wm = wave & 3;            // 4 waves down M
  const int wn = wave >> 2;           // 2 waves across O
  const int m0 = blockIdx.x * 128 + wm * 32;
  const int o0 = blockIdx.y * 128 + wn * 64;

  v8f acc[2][4];
#pragma unroll
  for (int i = 0; i < 2; ++i)
#pragma unroll
    for (int j = 0; j < 4; ++j) acc[i][j] = v8f{};

  for (int k0 = 0; k0 < Kdim; k0 += 32) {
    if (k0 + 32 < Kdim) {  // global_prefetch_b8 on CDNA5
      __builtin_prefetch(A + (size_t)m0 * lda + k0 + 32, 0, 1);
      __builtin_prefetch(W + (size_t)o0 * ldw + k0 + 32, 0, 1);
    }
    v16bf a[2], b[4];
#pragma unroll
    for (int i = 0; i < 2; ++i)
      a[i] = load_a_frag(A + (size_t)(m0 + 16 * i) * lda + k0, lda, lane);
#pragma unroll
    for (int j = 0; j < 4; ++j)
      b[j] = load_b_frag(W + (size_t)(o0 + 16 * j) * ldw + k0, ldw, lane);
#pragma unroll
    for (int i = 0; i < 2; ++i)
#pragma unroll
      for (int j = 0; j < 4; ++j)
        acc[i][j] = __builtin_amdgcn_wmma_f32_16x16x32_bf16(
            false, a[i], false, b[j], (short)0, acc[i][j], false, false);
  }

  // C/D layout: lanes 0-15 -> N=lane, M=r; lanes 16-31 -> N=lane-16, M=8+r.
  const int cl = lane & 15;
  const int rh = (lane >> 4) << 3;
#pragma unroll
  for (int i = 0; i < 2; ++i) {
#pragma unroll
    for (int j = 0; j < 4; ++j) {
#pragma unroll
      for (int r = 0; r < 8; ++r) {
        const int m = m0 + 16 * i + rh + r;
        const int o = o0 + 16 * j + cl;
        float v = acc[i][j][r] + bias[o];
        if (ACT) v = (v >= 0.f) ? v : 0.2f * v;
        if (OUT_BCN) {
          const int b_ = m / Nn, n_ = m - b_ * Nn;
          o_bcn[((size_t)b_ * Ochan + o) * Nn + n_] = v;
        } else {
          if (o_b1) o_b1[(size_t)m * ldo1 + o] = (bf16)v;
          if (o_b2) o_b2[(size_t)m * ldo2 + o] = (bf16)v;
          if (o_f)  o_f[(size_t)m * ldof + o] = v;
        }
      }
    }
  }
}

// ---------------------------------------------------------------------------
// Weight prep: fold BN (eval) scale into weight, produce bias; or plain cvt.
// ---------------------------------------------------------------------------
__global__ void fold_bn_weight(const float* __restrict__ W,
                               const float* __restrict__ g,
                               const float* __restrict__ b,
                               const float* __restrict__ m,
                               const float* __restrict__ v,
                               int O, int I,
                               bf16* __restrict__ Wb, float* __restrict__ bout) {
  const int idx = blockIdx.x * blockDim.x + threadIdx.x;
  if (idx >= O * I) return;
  const int o = idx / I;
  const float s = g[o] * rsqrtf(v[o] + 1e-5f);
  Wb[idx] = (bf16)(W[idx] * s);
  if (idx - o * I == 0) bout[o] = b[o] - m[o] * s;
}

__global__ void cvt_weight(const float* __restrict__ W, int n, bf16* __restrict__ Wb) {
  const int idx = blockIdx.x * blockDim.x + threadIdx.x;
  if (idx < n) Wb[idx] = (bf16)W[idx];
}

// [B,C,N] f32 -> token-major [T,C] bf16  (t = b*N + n)
__global__ void bcn_to_tc(const float* __restrict__ X, int C, int Nn, bf16* __restrict__ out) {
  const int idx = blockIdx.x * blockDim.x + threadIdx.x;
  const int total = BB * C * Nn;
  if (idx >= total) return;
  const int b = idx / (C * Nn);
  const int rem = idx - b * C * Nn;
  const int c = rem / Nn;
  const int n = rem - c * Nn;
  out[((size_t)b * Nn + n) * C + c] = (bf16)X[idx];
}

// ---------------------------------------------------------------------------
// q = x + sine positional embedding (computed inline), bf16 output.
// x is a column-concat of up to two f32 token buffers (widths w0, w1).
// ---------------------------------------------------------------------------
__global__ void q_add_pos(const float* __restrict__ x0, int w0,
                          const float* __restrict__ x1, int w1,
                          const float* __restrict__ xyz,  // [T,3]
                          int C, bf16* __restrict__ q) {
  const int idx = blockIdx.x * blockDim.x + threadIdx.x;
  if (idx >= TT * C) return;
  const int t = idx / C;
  const int c = idx - t * C;
  const int npf = (C / 3) / 2 * 2;
  float e = 0.f;
  if (c < 3 * npf) {
    const int a = c / npf;
    const int r = c - a * npf;
    const float expo = (float)(2 * (r / 2)) / (float)npf;
    const float freq = powf(10000.f, expo);
    const float pd = xyz[t * 3 + a] * 6.2831853071795864f / freq;
    e = (r & 1) ? cosf(pd) : sinf(pd);
  }
  const float xv = (c < w0) ? x0[(size_t)t * w0 + c] : x1[(size_t)t * w1 + (c - w0)];
  q[idx] = (bf16)(xv + e);
}

// tcomb = bf16( x - o1 + o2 ), x a column-concat of up to two f32 buffers.
__global__ void combine_residual(const float* __restrict__ x0, int w0,
                                 const float* __restrict__ x1, int w1,
                                 const float* __restrict__ o1,
                                 const float* __restrict__ o2,
                                 int C, bf16* __restrict__ out) {
  const int idx = blockIdx.x * blockDim.x + threadIdx.x;
  if (idx >= TT * C) return;
  const int t = idx / C;
  const int c = idx - t * C;
  const float xv = (c < w0) ? x0[(size_t)t * w0 + c] : x1[(size_t)t * w1 + (c - w0)];
  out[idx] = (bf16)(xv - o1[idx] + o2[idx]);
}

// ---------------------------------------------------------------------------
// KNN-masked attention, exploiting that softmax survives only on the <=16
// neighbor keys.  One wave per (b, n, head).  Shared probabilities applied to
// both V(src) and V(sim).  Duplicate neighbor indices are counted once
// (matches mask semantics: set(False) at a repeated index is idempotent).
// ---------------------------------------------------------------------------
__global__ __launch_bounds__(256) void attn_gather(
    const bf16* __restrict__ QK, int ldqk,   // [T, 2C]: Q cols 0..C-1, K cols C..2C-1
    const bf16* __restrict__ V1,
    const bf16* __restrict__ V2, int ldv,    // [T, C]
    const int* __restrict__ neigh,           // [B, N, 16]
    bf16* __restrict__ g1, bf16* __restrict__ g2, int ldg,
    int C) {
  const int lane = threadIdx.x & 31;
  const int wid = (blockIdx.x * blockDim.x + threadIdx.x) >> 5;
  if (wid >= TT * HH) return;
  const int h = wid % HH;
  const int t = wid / HH;
  const int b = t / NN;
  const int d = C / HH;          // 64 or 128
  const int dpl = d / 32;        // elems per lane: 2 or 4
  const float scale = rsqrtf((float)d);

  float qv[4];
#pragma unroll 4
  for (int i = 0; i < dpl; ++i)
    qv[i] = (float)QK[(size_t)t * ldqk + h * d + lane + 32 * i] * scale;

  int   idx[KK];
  float sc[KK];
  bool  valid[KK];
  const int* nb = neigh + (size_t)t * KK;
#pragma unroll
  for (int k = 0; k < KK; ++k) idx[k] = nb[k];
#pragma unroll
  for (int k = 0; k < KK; ++k) {
    bool ok = true;
    for (int kk = 0; kk < k; ++kk) ok = ok && (idx[kk] != idx[k]);
    valid[k] = ok;
  }

#pragma unroll
  for (int k = 0; k < KK; ++k) {
    const int tj = b * NN + idx[k];
    const bf16* kr = QK + (size_t)tj * ldqk + C + h * d;
    float p = 0.f;
    for (int i = 0; i < dpl; ++i) p += qv[i] * (float)kr[lane + 32 * i];
#pragma unroll
    for (int off = 16; off > 0; off >>= 1) p += __shfl_xor(p, off, 32);
    sc[k] = p;
  }

  float mx = -3.0e38f;
#pragma unroll
  for (int k = 0; k < KK; ++k) if (valid[k] && sc[k] > mx) mx = sc[k];
  float den = 0.f;
#pragma unroll
  for (int k = 0; k < KK; ++k) {
    sc[k] = valid[k] ? expf(sc[k] - mx) : 0.f;
    den += sc[k];
  }
  const float inv = 1.f / den;

  float a1[4] = {0.f, 0.f, 0.f, 0.f};
  float a2[4] = {0.f, 0.f, 0.f, 0.f};
#pragma unroll
  for (int k = 0; k < KK; ++k) {
    const float p = sc[k] * inv;
    const int tj = b * NN + idx[k];
    const bf16* v1r = V1 + (size_t)tj * ldv + h * d;
    const bf16* v2r = V2 + (size_t)tj * ldv + h * d;
    for (int i = 0; i < dpl; ++i) {
      a1[i] += p * (float)v1r[lane + 32 * i];
      a2[i] += p * (float)v2r[lane + 32 * i];
    }
  }
  for (int i = 0; i < dpl; ++i) {
    g1[(size_t)t * ldg + h * d + lane + 32 * i] = (bf16)a1[i];
    g2[(size_t)t * ldg + h * d + lane + 32 * i] = (bf16)a2[i];
  }
}

// ---------------------------------------------------------------------------
// Host-side orchestration
// ---------------------------------------------------------------------------
static inline void run_gemm(hipStream_t s, const bf16* A, int lda, const bf16* W, int ldw,
                            const float* bias, int Kdim, int Ochan,
                            bf16* ob1, int ldo1, bf16* ob2, int ldo2,
                            float* of, int ldof, float* obcn, int act) {
  dim3 grid(TT / 128, Ochan / 128), blk(256);
  if (obcn)
    gemm_wmma<1, 1><<<grid, blk, 0, s>>>(A, lda, W, ldw, bias, Kdim,
                                         nullptr, 0, nullptr, 0, nullptr, 0, obcn, Ochan, NN);
  else if (act)
    gemm_wmma<1, 0><<<grid, blk, 0, s>>>(A, lda, W, ldw, bias, Kdim,
                                         ob1, ldo1, ob2, ldo2, of, ldof, nullptr, Ochan, NN);
  else
    gemm_wmma<0, 0><<<grid, blk, 0, s>>>(A, lda, W, ldw, bias, Kdim,
                                         ob1, ldo1, ob2, ldo2, of, ldof, nullptr, Ochan, NN);
}

extern "C" void kernel_launch(void* const* d_in, const int* in_sizes, int n_in,
                              void* d_out, int out_size, void* d_ws, size_t ws_size,
                              hipStream_t stream) {
  (void)in_sizes; (void)n_in; (void)out_size; (void)ws_size;
  // ---- inputs (setup_inputs dict order, params flattened depth-first) ----
  const float* src   = (const float*)d_in[0];   // [B,512,N]
  const float* simm  = (const float*)d_in[1];   // [B,1024,N]
  const float* xyz   = (const float*)d_in[2];   // [B,N,3]
  const int*   neigh = (const int*)d_in[3];     // [B,N,16]
  const float* p1_w = (const float*)d_in[4],  *p1_g = (const float*)d_in[5],
             * p1_b = (const float*)d_in[6],  *p1_m = (const float*)d_in[7],
             * p1_v = (const float*)d_in[8];
  const float* p21_w = (const float*)d_in[9],  *p21_g = (const float*)d_in[10],
             * p21_b = (const float*)d_in[11], *p21_m = (const float*)d_in[12],
             * p21_v = (const float*)d_in[13];
  const float* p22_w = (const float*)d_in[14], *p22_g = (const float*)d_in[15],
             * p22_b = (const float*)d_in[16], *p22_m = (const float*)d_in[17],
             * p22_v = (const float*)d_in[18];
  const float* l0_inw = (const float*)d_in[19], *l0_inb = (const float*)d_in[20],
             * l0_ow  = (const float*)d_in[21], *l0_ob  = (const float*)d_in[22],
             * l0_lw  = (const float*)d_in[23], *l0_g   = (const float*)d_in[24],
             * l0_bb  = (const float*)d_in[25], *l0_m   = (const float*)d_in[26],
             * l0_v   = (const float*)d_in[27];
  const float* l1_inw = (const float*)d_in[28], *l1_inb = (const float*)d_in[29],
             * l1_ow  = (const float*)d_in[30], *l1_ob  = (const float*)d_in[31],
             * l1_lw  = (const float*)d_in[32], *l1_g   = (const float*)d_in[33],
             * l1_bb  = (const float*)d_in[34], *l1_m   = (const float*)d_in[35],
             * l1_v   = (const float*)d_in[36];
  const float* mp_w = (const float*)d_in[37], *mp_g = (const float*)d_in[38],
             * mp_b = (const float*)d_in[39], *mp_m = (const float*)d_in[40],
             * mp_v = (const float*)d_in[41];

  // ---- workspace bump allocator ----
  uintptr_t cur = (uintptr_t)d_ws;
  auto alloc = [&](size_t bytes) -> void* {
    cur = (cur + 255) & ~(uintptr_t)255;
    void* r = (void*)cur;
    cur += bytes;
    return r;
  };
  auto abf = [&](size_t n) { return (bf16*)alloc(n * sizeof(bf16)); };
  auto af  = [&](size_t n) { return (float*)alloc(n * sizeof(float)); };

  // folded / converted weights (bf16) + folded biases (f32)
  bf16* wp1  = abf(512 * 512);     float* bp1  = af(512);
  bf16* wp21 = abf(512 * 1024);    float* bp21 = af(512);
  bf16* wp22 = abf(1024 * 1024);   float* bp22 = af(1024);
  bf16* wl0  = abf(512 * 512);     float* bl0  = af(512);    // lin0 (+bn0)
  bf16* wl1  = abf(1024 * 1024);   float* bl1  = af(1024);   // lin1 (+bn1)
  bf16* wmp  = abf(512 * 1536);    float* bmp  = af(512);
  bf16* win0 = abf(1536 * 512);
  bf16* wout0 = abf(512 * 512);
  bf16* win1 = abf(3072 * 1024);
  bf16* wout1 = abf(1024 * 1024);

  // activations (token-major [T, C])
  bf16* srcT = abf((size_t)TT * 512);
  bf16* simT = abf((size_t)TT * 1024);
  bf16* sim1 = abf((size_t)TT * 512);
  bf16* sim2 = abf((size_t)TT * 1024);
  bf16* X1   = abf((size_t)TT * 1024);   // [x0 | y0]
  bf16* F    = abf((size_t)TT * 1536);   // [y0 | y1]
  float* x0f = af((size_t)TT * 512);
  float* y0f = af((size_t)TT * 512);
  bf16* qb   = abf((size_t)TT * 1024);
  bf16* QK   = abf((size_t)TT * 2048);
  bf16* Vb1  = abf((size_t)TT * 1024);
  bf16* Vb2  = abf((size_t)TT * 1024);
  bf16* g1   = abf((size_t)TT * 1024);
  bf16* g2   = abf((size_t)TT * 1024);
  float* o1f = af((size_t)TT * 1024);
  float* o2f = af((size_t)TT * 1024);
  bf16* tcb  = abf((size_t)TT * 1024);

  const int TPB = 256;
  auto blocks = [&](long n) { return (int)((n + TPB - 1) / TPB); };

  // ---- weight prep ----
  fold_bn_weight<<<blocks(512 * 512), TPB, 0, stream>>>(p1_w, p1_g, p1_b, p1_m, p1_v, 512, 512, wp1, bp1);
  fold_bn_weight<<<blocks(512 * 1024), TPB, 0, stream>>>(p21_w, p21_g, p21_b, p21_m, p21_v, 512, 1024, wp21, bp21);
  fold_bn_weight<<<blocks(1024 * 1024), TPB, 0, stream>>>(p22_w, p22_g, p22_b, p22_m, p22_v, 1024, 1024, wp22, bp22);
  fold_bn_weight<<<blocks(512 * 512), TPB, 0, stream>>>(l0_lw, l0_g, l0_bb, l0_m, l0_v, 512, 512, wl0, bl0);
  fold_bn_weight<<<blocks(1024 * 1024), TPB, 0, stream>>>(l1_lw, l1_g, l1_bb, l1_m, l1_v, 1024, 1024, wl1, bl1);
  fold_bn_weight<<<blocks(512 * 1536), TPB, 0, stream>>>(mp_w, mp_g, mp_b, mp_m, mp_v, 512, 1536, wmp, bmp);
  cvt_weight<<<blocks(1536 * 512), TPB, 0, stream>>>(l0_inw, 1536 * 512, win0);
  cvt_weight<<<blocks(512 * 512), TPB, 0, stream>>>(l0_ow, 512 * 512, wout0);
  cvt_weight<<<blocks(3072 * 1024), TPB, 0, stream>>>(l1_inw, 3072 * 1024, win1);
  cvt_weight<<<blocks(1024 * 1024), TPB, 0, stream>>>(l1_ow, 1024 * 1024, wout1);

  // ---- tokenize inputs ----
  bcn_to_tc<<<blocks((long)BB * 512 * NN), TPB, 0, stream>>>(src, 512, NN, srcT);
  bcn_to_tc<<<blocks((long)BB * 1024 * NN), TPB, 0, stream>>>(simm, 1024, NN, simT);

  // ---- pre-MLPs ----
  run_gemm(stream, srcT, 512, wp1, 512, bp1, 512, 512, X1, 1024, nullptr, 0, x0f, 512, nullptr, 1);
  run_gemm(stream, simT, 1024, wp21, 1024, bp21, 1024, 512, sim1, 512, nullptr, 0, nullptr, 0, nullptr, 1);
  run_gemm(stream, simT, 1024, wp22, 1024, bp22, 1024, 1024, sim2, 1024, nullptr, 0, nullptr, 0, nullptr, 1);

  // ================= layer 0 (C = 512) =================
  {
    const int C = 512;
    q_add_pos<<<blocks((long)TT * C), TPB, 0, stream>>>(x0f, 512, nullptr, 0, xyz, C, qb);
    // Q|K from q  (in_w rows 0..2C)
    run_gemm(stream, qb, C, win0, C, l0_inb, C, 2 * C, QK, 2 * C, nullptr, 0, nullptr, 0, nullptr, 0);
    // V(src) / V(sim) from Wv = in_w rows 2C..3C
    run_gemm(stream, X1, 1024, win0 + (size_t)2 * C * C, C, l0_inb + 2 * C, C, C, Vb1, C, nullptr, 0, nullptr, 0, nullptr, 0);
    run_gemm(stream, sim1, C, win0 + (size_t)2 * C * C, C, l0_inb + 2 * C, C, C, Vb2, C, nullptr, 0, nullptr, 0, nullptr, 0);
    attn_gather<<<(TT * HH * 32) / TPB, TPB, 0, stream>>>(QK, 2 * C, Vb1, Vb2, C, neigh, g1, g2, C, C);
    run_gemm(stream, g1, C, wout0, C, l0_ob, C, C, nullptr, 0, nullptr, 0, o1f, C, nullptr, 0);
    run_gemm(stream, g2, C, wout0, C, l0_ob, C, C, nullptr, 0, nullptr, 0, o2f, C, nullptr, 0);
    combine_residual<<<blocks((long)TT * C), TPB, 0, stream>>>(x0f, 512, nullptr, 0, o1f, o2f, C, tcb);
    // y0 = leaky(bn(lin(t)))  ->  X1[:,512:], F[:,0:512], y0f
    run_gemm(stream, tcb, C, wl0, C, bl0, C, C, X1 + 512, 1024, F, 1536, y0f, 512, nullptr, 1);
  }

  // ================= layer 1 (C = 1024) =================
  {
    const int C = 1024;
    q_add_pos<<<blocks((long)TT * C), TPB, 0, stream>>>(x0f, 512, y0f, 512, xyz, C, qb);
    run_gemm(stream, qb, C, win1, C, l1_inb, C, 2 * C, QK, 2 * C, nullptr, 0, nullptr, 0, nullptr, 0);
    run_gemm(stream, X1, 1024, win1 + (size_t)2 * C * C, C, l1_inb + 2 * C, C, C, Vb1, C, nullptr, 0, nullptr, 0, nullptr, 0);
    run_gemm(stream, sim2, C, win1 + (size_t)2 * C * C, C, l1_inb + 2 * C, C, C, Vb2, C, nullptr, 0, nullptr, 0, nullptr, 0);
    attn_gather<<<(TT * HH * 32) / TPB, TPB, 0, stream>>>(QK, 2 * C, Vb1, Vb2, C, neigh, g1, g2, C, C);
    run_gemm(stream, g1, C, wout1, C, l1_ob, C, C, nullptr, 0, nullptr, 0, o1f, C, nullptr, 0);
    run_gemm(stream, g2, C, wout1, C, l1_ob, C, C, nullptr, 0, nullptr, 0, o2f, C, nullptr, 0);
    combine_residual<<<blocks((long)TT * C), TPB, 0, stream>>>(x0f, 512, y0f, 512, o1f, o2f, C, tcb);
    // y1 -> F[:,512:1536]
    run_gemm(stream, tcb, C, wl1, C, bl1, C, C, F + 512, 1536, nullptr, 0, nullptr, 0, nullptr, 1);
  }

  // ---- final mlp over concat(inter) = F [T,1536] -> d_out [B,512,N] ----
  run_gemm(stream, F, 1536, wmp, 1536, bmp, 1536, 512,
           nullptr, 0, nullptr, 0, nullptr, 0, (float*)d_out, 1);
}
